// Ponita_20134806684148
// MI455X (gfx1250) — matrix operand order repaired and checked by reading, not compiled, not run on hardware
//
#include <hip/hip_runtime.h>
#include <hip/hip_bf16.h>

#define NN    10000
#define EE    80000
#define OO    10
#define CC    64
#define DIN   16
#define OUTC  16
#define LL    2
#define HID   256   // WIDEN * C

typedef __attribute__((ext_vector_type(16))) __bf16 v16bf;
typedef __attribute__((ext_vector_type(8)))  float  f32x8;

union AB16 { v16bf v; unsigned short u[16]; };   // one WMMA fragment: 32B per lane

__device__ __forceinline__ unsigned short f2bf(float f) {
  unsigned int u = __float_as_uint(f);
  u += 0x7FFFu + ((u >> 16) & 1u);   // round-to-nearest-even truncation
  return (unsigned short)(u >> 16);
}

// branchless tanh-gelu: 0.5x(1+tanh(z)) == x*sigmoid(2z),  z = 0.79788456(x+0.044715x^3)
__device__ __forceinline__ float gelu_f(float x) {
  float z = 1.5957691216f * (x + 0.044715f * x * x * x);
  return x / (1.0f + __expf(-z));
}

__device__ __forceinline__ f32x8 wmma_bf16(const AB16& a, const AB16& b, f32x8 c) {
  return __builtin_amdgcn_wmma_f32_16x16x32_bf16(false, a.v, false, b.v, (short)0, c,
                                                 false, false);
}

// inverse fragment maps (position = lane*16 + slot within one 512-u16 tile)
__device__ __forceinline__ int a_pos(int row, int k) {      // A tile 16 rows x 32 K
  return (row + ((k >> 3) & 1) * 16) * 16 + ((k & 7) | ((k & 16) >> 1));
}
__device__ __forceinline__ int b_pos(int k, int col) {      // B tile 32 K x 16 cols
  return (col + ((k >> 3) & 1) * 16) * 16 + ((k & 7) | ((k & 16) >> 1));
}

__device__ __forceinline__ float wave_sum32(float v) {
#pragma unroll
  for (int off = 16; off > 0; off >>= 1) v += __shfl_xor(v, off, 32);
  return v;
}

__device__ __forceinline__ void fib_ori(int k, float* v) {
  float i = (float)k + 0.5f;
  float phi = acosf(1.0f - 2.0f * i / (float)OO);
  float theta = 3.14159265358979323846f * (1.0f + 2.2360679774997896964f) * i;
  float sp = sinf(phi);
  v[0] = cosf(theta) * sp; v[1] = sinf(theta) * sp; v[2] = cosf(phi);
}

// ---------------------------------------------------------------- precompute
__global__ void ponita_pre(const float* __restrict__ fbw1, const float* __restrict__ fbb1,
                           const float* __restrict__ fbw2, const float* __restrict__ fbb2,
                           const float* __restrict__ conv_fw,
                           float* __restrict__ ori, float* __restrict__ fk) {
  int t = threadIdx.x;
  if (t < OO) { float v[3]; fib_ori(t, v); ori[t*3] = v[0]; ori[t*3+1] = v[1]; ori[t*3+2] = v[2]; }
  if (t < OO * OO) {
    int p = t / OO, o = t - p * OO;
    float vp[3], vo[3];
    fib_ori(p, vp); fib_ori(o, vo);
    float d = vp[0]*vo[0] + vp[1]*vo[1] + vp[2]*vo[2];
    float pf[4] = { d, d*d, d*d*d, d*d*d*d };
    float f1[CC], f2[CC];
    for (int c = 0; c < CC; ++c) {
      float a = fbb1[c];
      for (int j = 0; j < 4; ++j) a += pf[j] * fbw1[j * CC + c];
      f1[c] = gelu_f(a);
    }
    for (int c = 0; c < CC; ++c) {
      float a = fbb2[c];
      for (int j = 0; j < CC; ++j) a += f1[j] * fbw2[j * CC + c];
      f2[c] = gelu_f(a);
    }
    for (int l = 0; l < LL; ++l)
      for (int c = 0; c < CC; ++c) {
        float a = 0.f;
        for (int bd = 0; bd < CC; ++bd) a += f2[bd] * conv_fw[l * CC * CC + bd * CC + c];
        fk[l * OO * OO * CC + t * CC + c] = a;
      }
  }
}

// ---------------------------------------------------------------- embed
__global__ void ponita_embed(const float* __restrict__ x, const float* __restrict__ w,
                             float* __restrict__ h) {
  int idx = blockIdx.x * blockDim.x + threadIdx.x;
  if (idx >= NN * CC) return;
  int n = idx / CC, c = idx - n * CC;
  float a = 0.f;
#pragma unroll
  for (int d = 0; d < DIN; ++d) a += x[n * DIN + d] * w[d * CC + c];
#pragma unroll
  for (int o = 0; o < OO; ++o) h[(n * OO + o) * CC + c] = a;
}

// ---------------------------------------------------------------- kernel basis (E*O rows)
// Produces kb in *global A-fragment layout*: [tile][kblk(2)][lane(32)][slot(16)] bf16.
__global__ void __launch_bounds__(256) ponita_kb(
    const float* __restrict__ pos, const int* __restrict__ ei, const float* __restrict__ ori,
    const float* __restrict__ w1, const float* __restrict__ b1,
    const float* __restrict__ w2, const float* __restrict__ b2,
    unsigned short* __restrict__ kb) {
  __shared__ AB16 w1B[4 * 32];        // 32x64  -> 4 B tiles
  __shared__ AB16 w2B[8 * 32];        // 64x64  -> 8 B tiles
  __shared__ float b1s[CC], b2s[CC];
  __shared__ AB16 polyA[8][32];       // per wave: 16x32 A tile
  __shared__ AB16 t1A[8][2 * 32];     // per wave: 16x64 -> 2 A tiles
  int tid = threadIdx.x;
  unsigned short* w1Bu = (unsigned short*)w1B;
  unsigned short* w2Bu = (unsigned short*)w2B;
  for (int i = tid; i < 32 * CC; i += 256) {
    int k = i >> 6, n = i & 63;
    w1Bu[(n >> 4) * 512 + b_pos(k, n & 15)] = (k < 30) ? f2bf(w1[i]) : (unsigned short)0;
  }
  for (int i = tid; i < CC * CC; i += 256) {
    int k = i >> 6, n = i & 63;
    w2Bu[((k >> 5) * 4 + (n >> 4)) * 512 + b_pos(k & 31, n & 15)] = f2bf(w2[i]);
  }
  if (tid < CC) { b1s[tid] = b1[tid]; b2s[tid] = b2[tid]; }
  __syncthreads();

  int wid = tid >> 5, lane = tid & 31;
  int tile = blockIdx.x * 8 + wid;                  // 50000 tiles of 16 (e,o) rows
  unsigned short* pAu = (unsigned short*)polyA[wid];
  unsigned short* tAu = (unsigned short*)t1A[wid];

  if (lane < 16) {
    int r = tile * 16 + lane;
    int e = r / OO, o = r - e * OO;
    int s = ei[e], rc = ei[EE + e];
    float rx = pos[s*3+0] - pos[rc*3+0];
    float ry = pos[s*3+1] - pos[rc*3+1];
    float rz = pos[s*3+2] - pos[rc*3+2];
    float ox = ori[o*3+0], oy = ori[o*3+1], oz = ori[o*3+2];
    float i1 = rx*ox + ry*oy + rz*oz;
    float px = rx - i1*ox, py = ry - i1*oy, pz = rz - i1*oz;
    float i2 = sqrtf(px*px + py*py + pz*pz);
    float feats[30];
    feats[0] = i1; feats[1] = i2;
    int curStart = 0, curLen = 2, outPos = 2;
#pragma unroll
    for (int dg = 0; dg < 3; ++dg) {
#pragma unroll 16
      for (int i = 0; i < curLen; ++i) {
        feats[outPos + 2*i]     = feats[curStart + i] * i1;
        feats[outPos + 2*i + 1] = feats[curStart + i] * i2;
      }
      curStart = outPos; outPos += 2 * curLen; curLen *= 2;
    }
#pragma unroll
    for (int k = 0; k < 30; ++k) pAu[a_pos(lane, k)] = f2bf(feats[k]);
    pAu[a_pos(lane, 30)] = 0; pAu[a_pos(lane, 31)] = 0;
  }
  __syncthreads();

  int hf = lane >> 4, cn = lane & 15;
  {   // GEMM1: poly[16x32] @ w1[32x64], gelu -> t1A (A-fragment layout)
    AB16 a = polyA[wid][lane];
#pragma unroll
    for (int ct = 0; ct < 4; ++ct) {
      f32x8 acc = {};
      acc = wmma_bf16(a, w1B[ct * 32 + lane], acc);
#pragma unroll
      for (int vg = 0; vg < 8; ++vg) {
        int m = vg + 8 * hf, col = ct * 16 + cn;
        tAu[(col >> 5) * 512 + a_pos(m, col & 31)] = f2bf(gelu_f(acc[vg] + b1s[col]));
      }
    }
  }
  __syncthreads();
  {   // GEMM2: t1[16x64] @ w2[64x64], gelu -> kb global (A-fragment layout)
    AB16 a0 = t1A[wid][lane];
    AB16 a1 = t1A[wid][32 + lane];
#pragma unroll
    for (int ct = 0; ct < 4; ++ct) {
      f32x8 acc = {};
      acc = wmma_bf16(a0, w2B[ct * 32 + lane], acc);
      acc = wmma_bf16(a1, w2B[(4 + ct) * 32 + lane], acc);
#pragma unroll
      for (int vg = 0; vg < 8; ++vg) {
        int m = vg + 8 * hf, col = ct * 16 + cn;
        kb[(size_t)tile * 1024 + (col >> 5) * 512 + a_pos(m, col & 31)] =
            f2bf(gelu_f(acc[vg] + b2s[col]));
      }
    }
  }
}

// ---------------------------------------------------------------- per-layer message+scatter
__global__ void __launch_bounds__(256) ponita_msg(
    const unsigned short* __restrict__ kb, const float* __restrict__ kwl,
    const int* __restrict__ ei, const float* __restrict__ h, float* __restrict__ h1) {
  __shared__ AB16 kwB[8 * 32];        // 64x64 -> 8 B tiles
  int tid = threadIdx.x;
  unsigned short* kwBu = (unsigned short*)kwB;
  for (int i = tid; i < CC * CC; i += 256) {
    int k = i >> 6, n = i & 63;
    kwBu[((k >> 5) * 4 + (n >> 4)) * 512 + b_pos(k & 31, n & 15)] = f2bf(kwl[i]);
  }
  __syncthreads();
  int wid = tid >> 5, lane = tid & 31;
  int tile = blockIdx.x * 8 + wid;
  const AB16* kA = (const AB16*)(kb + (size_t)tile * 1024);
  if (tile + 8 < (EE * OO) / 16)
    __builtin_prefetch(kb + (size_t)(tile + 8) * 1024, 0, 1);   // global_prefetch_b8
  AB16 a0 = kA[lane];            // coalesced 32B/lane global_load_b128 pairs
  AB16 a1 = kA[32 + lane];
  int hf = lane >> 4, cn = lane & 15;
  int sbase[8], rbase[8];
#pragma unroll
  for (int vg = 0; vg < 8; ++vg) {
    int r = tile * 16 + vg + 8 * hf;
    int e = r / OO, o = r - e * OO;
    sbase[vg] = (ei[e] * OO + o) * CC;
    rbase[vg] = (ei[EE + e] * OO + o) * CC;
  }
#pragma unroll
  for (int ct = 0; ct < 4; ++ct) {
    f32x8 acc = {};
    acc = wmma_bf16(a0, kwB[ct * 32 + lane], acc);
    acc = wmma_bf16(a1, kwB[(4 + ct) * 32 + lane], acc);
#pragma unroll
    for (int vg = 0; vg < 8; ++vg) {
      int col = ct * 16 + cn;
      float hv = h[sbase[vg] + col];
      atomicAdd(&h1[rbase[vg] + col], hv * acc[vg]);
    }
  }
}

// ---------------------------------------------------------------- per-layer fiber conv + MLP + readout
__global__ void __launch_bounds__(256) ponita_fiber(
    const float* __restrict__ h1, const float* __restrict__ fkl,
    const float* __restrict__ cb, const float* __restrict__ ns, const float* __restrict__ nbv,
    const float* __restrict__ w1, const float* __restrict__ b1,
    const float* __restrict__ w2, const float* __restrict__ b2,
    const float* __restrict__ row_, const float* __restrict__ rob,
    float* __restrict__ h, float* __restrict__ nodeOut) {
  extern __shared__ AB16 smemv[];
  AB16* w1B = smemv;                  // 64x256 -> 32 B tiles (kblk*16 + cblk)
  AB16* w2B = w1B + 32 * 32;          // 256x64 -> 32 B tiles (kblk*4 + cblk)
  int tid = threadIdx.x, wid = tid >> 5, lane = tid & 31;
  AB16* A1  = w2B + 32 * 32 + wid * (2 * 32 + 8 * 32);
  AB16* mid = A1 + 2 * 32;
  unsigned short* w1Bu = (unsigned short*)w1B;
  unsigned short* w2Bu = (unsigned short*)w2B;
  unsigned short* A1u  = (unsigned short*)A1;
  unsigned short* midu = (unsigned short*)mid;

  for (int i = tid; i < CC * HID; i += 256) {
    int k = i >> 8, n = i & 255;
    w1Bu[((k >> 5) * 16 + (n >> 4)) * 512 + b_pos(k & 31, n & 15)] = f2bf(w1[i]);
  }
  for (int i = tid; i < HID * CC; i += 256) {
    int k = i >> 6, n = i & 63;
    w2Bu[((k >> 5) * 4 + (n >> 4)) * 512 + b_pos(k & 31, n & 15)] = f2bf(w2[i]);
  }
  __syncthreads();

  int n = blockIdx.x * 8 + wid;
  int c0 = lane, c1 = lane + 32;
  // fiber (orientation) conv + layernorm -> bf16 A fragments (rows 10..15 zero)
  for (int p = 0; p < OO; ++p) {
    float a0 = 0.f, a1 = 0.f;
#pragma unroll
    for (int o = 0; o < OO; ++o) {
      const float* hr = h1 + (size_t)(n * OO + o) * CC;
      const float* fr = fkl + (p * OO + o) * CC;
      a0 += hr[c0] * fr[c0];
      a1 += hr[c1] * fr[c1];
    }
    a0 = a0 * (1.0f / OO) + cb[c0];
    a1 = a1 * (1.0f / OO) + cb[c1];
    float mu = wave_sum32(a0 + a1) * (1.0f / CC);
    float d0 = a0 - mu, d1 = a1 - mu;
    float var = wave_sum32(d0 * d0 + d1 * d1) * (1.0f / CC);
    float rs = rsqrtf(var + 1e-6f);
    A1u[a_pos(p, lane)]       = f2bf(d0 * rs * ns[c0] + nbv[c0]);   // kblk 0
    A1u[512 + a_pos(p, lane)] = f2bf(d1 * rs * ns[c1] + nbv[c1]);   // kblk 1
  }
#pragma unroll
  for (int p = OO; p < 16; ++p) { A1u[a_pos(p, lane)] = 0; A1u[512 + a_pos(p, lane)] = 0; }
  __syncthreads();

  int hf = lane >> 4, cn = lane & 15;
  {   // GEMM1: [16x64] @ [64x256], gelu -> mid (A-fragment layout)
    AB16 a0 = A1[lane];
    AB16 a1 = A1[32 + lane];
#pragma unroll
    for (int ct = 0; ct < 16; ++ct) {
      f32x8 acc = {};
      acc = wmma_bf16(a0, w1B[ct * 32 + lane], acc);
      acc = wmma_bf16(a1, w1B[(16 + ct) * 32 + lane], acc);
#pragma unroll
      for (int vg = 0; vg < 8; ++vg) {
        int m = vg + 8 * hf, col = ct * 16 + cn;
        midu[(col >> 5) * 512 + a_pos(m, col & 31)] = f2bf(gelu_f(acc[vg] + b1[col]));
      }
    }
  }
  __syncthreads();
  {   // GEMM2: [16x256] @ [256x64] -> residual into h
    AB16 at[8];
#pragma unroll
    for (int ks = 0; ks < 8; ++ks) at[ks] = mid[ks * 32 + lane];
#pragma unroll
    for (int ct = 0; ct < 4; ++ct) {
      f32x8 acc = {};
#pragma unroll
      for (int ks = 0; ks < 8; ++ks)
        acc = wmma_bf16(at[ks], w2B[(ks * 4 + ct) * 32 + lane], acc);
#pragma unroll
      for (int vg = 0; vg < 8; ++vg) {
        int m = vg + 8 * hf, col = ct * 16 + cn;
        if (m < OO) {
          size_t idx = (size_t)(n * OO + m) * CC + col;
          h[idx] += acc[vg] + b2[col];
        }
      }
    }
  }
  // readout: sum over orientations of h_new @ ro_w + ro_b, accumulated per node
  int j = lane & 15, ch = lane >> 4;
  float s = 0.f;
  for (int p = 0; p < OO; ++p) {
    const float* hp = h + (size_t)(n * OO + p) * CC;
#pragma unroll
    for (int c = 0; c < 32; ++c) s += hp[ch * 32 + c] * row_[(ch * 32 + c) * OUTC + j];
  }
  s += __shfl_xor(s, 16, 32);
  if (lane < 16) nodeOut[n * OUTC + j] += s + (float)OO * rob[j];
}

// ---------------------------------------------------------------- batch pooling
__global__ void ponita_pool(const float* __restrict__ nodeOut, const int* __restrict__ batch,
                            float* __restrict__ out) {
  int idx = blockIdx.x * blockDim.x + threadIdx.x;
  if (idx >= NN * OUTC) return;
  int n = idx / OUTC, j = idx - n * OUTC;
  atomicAdd(&out[batch[n] * OUTC + j], nodeOut[idx] * (1.0f / (OO * LL)));
}

// ---------------------------------------------------------------- launch
extern "C" void kernel_launch(void* const* d_in, const int* in_sizes, int n_in,
                              void* d_out, int out_size, void* d_ws, size_t ws_size,
                              hipStream_t stream) {
  (void)in_sizes; (void)n_in; (void)ws_size;
  const float* x        = (const float*)d_in[0];
  const float* pos      = (const float*)d_in[1];
  const float* basis_w1 = (const float*)d_in[2];
  const float* basis_b1 = (const float*)d_in[3];
  const float* basis_w2 = (const float*)d_in[4];
  const float* basis_b2 = (const float*)d_in[5];
  const float* fbw1     = (const float*)d_in[6];
  const float* fbb1     = (const float*)d_in[7];
  const float* fbw2     = (const float*)d_in[8];
  const float* fbb2     = (const float*)d_in[9];
  const float* embed_w  = (const float*)d_in[10];
  const float* conv_kw  = (const float*)d_in[11];
  const float* conv_fw  = (const float*)d_in[12];
  const float* conv_b   = (const float*)d_in[13];
  const float* norm_s   = (const float*)d_in[14];
  const float* norm_b   = (const float*)d_in[15];
  const float* lin1_w   = (const float*)d_in[16];
  const float* lin1_b   = (const float*)d_in[17];
  const float* lin2_w   = (const float*)d_in[18];
  const float* lin2_b   = (const float*)d_in[19];
  const float* ro_w     = (const float*)d_in[20];
  const float* ro_b     = (const float*)d_in[21];
  const int*   edge_idx = (const int*)d_in[22];
  const int*   batch    = (const int*)d_in[23];
  float* out = (float*)d_out;

  float* wsf = (float*)d_ws;
  float* ori = wsf;                                          // 32
  float* fk  = wsf + 32;                                     // LL*OO*OO*CC
  float* h   = fk + LL * OO * OO * CC;                       // NN*OO*CC
  float* h1  = h + (size_t)NN * OO * CC;                     // NN*OO*CC
  float* nodeOut = h1 + (size_t)NN * OO * CC;                // NN*OUTC
  unsigned short* kb = (unsigned short*)(nodeOut + NN * OUTC); // EE*OO*CC bf16, A-fragment layout

  hipMemsetAsync(nodeOut, 0, sizeof(float) * NN * OUTC, stream);
  hipMemsetAsync(d_out, 0, sizeof(float) * out_size, stream);

  ponita_pre<<<1, 128, 0, stream>>>(fbw1, fbb1, fbw2, fbb2, conv_fw, ori, fk);
  ponita_embed<<<(NN * CC + 255) / 256, 256, 0, stream>>>(x, embed_w, h);
  ponita_kb<<<(EE * OO) / 16 / 8, 256, 0, stream>>>(pos, edge_idx, ori,
                                                    basis_w1, basis_b1, basis_w2, basis_b2, kb);

  const int FIBER_SMEM = (32 * 32 + 32 * 32 + 8 * (2 * 32 + 8 * 32)) * (int)sizeof(AB16); // 147456 B
  (void)hipFuncSetAttribute((const void*)ponita_fiber,
                            hipFuncAttributeMaxDynamicSharedMemorySize, FIBER_SMEM);

  for (int l = 0; l < LL; ++l) {
    hipMemsetAsync(h1, 0, sizeof(float) * (size_t)NN * OO * CC, stream);
    ponita_msg<<<(EE * OO) / 16 / 8, 256, 0, stream>>>(kb, conv_kw + l * CC * CC,
                                                       edge_idx, h, h1);
    ponita_fiber<<<NN / 8, 256, FIBER_SMEM, stream>>>(
        h1, fk + l * OO * OO * CC, conv_b + l * CC, norm_s + l * CC, norm_b + l * CC,
        lin1_w + l * CC * HID, lin1_b + l * HID, lin2_w + l * HID * CC, lin2_b + l * CC,
        ro_w + l * CC * OUTC, ro_b + l * OUTC, h, nodeOut);
  }
  ponita_pool<<<(NN * OUTC + 255) / 256, 256, 0, stream>>>(nodeOut, batch, out);
}
